// Attention_FishPP_67937792688803
// MI455X (gfx1250) — matrix operand order repaired and checked by reading, not compiled, not run on hardware
//
#include <hip/hip_runtime.h>
#include <hip/hip_bf16.h>

// ---------------- problem constants ----------------
#define BB     64
#define NN     197
#define CC     768
#define HH     12
#define GHN    2
#define HRN    6
#define DD     64
#define THEADS 16            // TOTAL_HEADS
#define MROWS  (BB * NN)     // 12608 (== 788*16 exactly)
#define MT788  (MROWS / 16)  // 788 (even -> 394 32-row tiles)
#define NPAD   224           // 197 padded to 7 * 32 for K-chunks
#define MTILES 13            // ceil(197/16)
#define SCALE  0.125f        // 64^-0.5

// ---------------- WMMA types ----------------
typedef __attribute__((ext_vector_type(16))) __bf16 v16bf;
typedef __attribute__((ext_vector_type(8)))  float  v8f;

union FragA { v16bf v; uint4 u[2]; };

__device__ __forceinline__ v8f wmma_bf16(v16bf a, v16bf b, v8f c) {
  // (neg_a, A, neg_b, B, c_mod, C, reuse_a, reuse_b)
  return __builtin_amdgcn_wmma_f32_16x16x32_bf16(false, a, false, b, (short)0, c,
                                                 false, false);
}

// A fragment (16-bit, 16x32): lane = row (lane%16), K interleaved by lane-half
__device__ __forceinline__ void load_fragA(FragA& f, const __bf16* row, int lh) {
  f.u[0] = *(const uint4*)(row + lh * 8);
  f.u[1] = *(const uint4*)(row + 16 + lh * 8);
}
// B fragment (16-bit, 32x16) from B^T row-major: lane = out col, K interleaved by lane-half
__device__ __forceinline__ void load_fragB(FragA& f, const __bf16* row, int lh) {
  f.u[0] = *(const uint4*)(row + lh * 8);
  f.u[1] = *(const uint4*)(row + 16 + lh * 8);
}

// ---------------- prep kernels ----------------
__global__ void k_f32_to_bf16(const float* __restrict__ in,
                              __bf16* __restrict__ out, int n) {
  int i = blockIdx.x * blockDim.x + threadIdx.x;
  if (i < n) out[i] = (__bf16)in[i];
}

// in: (rows x cols) fp32 row-major  ->  out: (cols x rows) bf16 row-major
__global__ void k_transpose_bf16(const float* __restrict__ in,
                                 __bf16* __restrict__ out, int rows, int cols) {
  int i = blockIdx.x * blockDim.x + threadIdx.x;
  if (i >= rows * cols) return;
  int c = i / rows, k = i - c * rows;
  out[(size_t)c * rows + k] = (__bf16)in[(size_t)k * cols + c];
}

// mw(n,m,h) = masks(n,m,:) @ mask_proj(:,h) + mask_base(h)
__global__ void k_maskw(const float* __restrict__ masks,
                        const float* __restrict__ mproj,
                        const float* __restrict__ mbase,
                        float* __restrict__ mw) {
  int i = blockIdx.x * blockDim.x + threadIdx.x;  // over N*N
  if (i >= NN * NN) return;
  float l0 = masks[i * 3 + 0], l1 = masks[i * 3 + 1], l2 = masks[i * 3 + 2];
#pragma unroll
  for (int h = 0; h < HH; ++h)
    mw[(size_t)i * HH + h] =
        l0 * mproj[0 * HH + h] + l1 * mproj[1 * HH + h] + l2 * mproj[2 * HH + h] + mbase[h];
}

// ---------------- QKV GEMM: (12608 x 768) @ (768 x 1024) + bias -------------
// 2x2 register blocking: one 32x32 output tile per wave (2 loads per WMMA).
// Scatter D into q(B,GH,N,64), k(B,GH,N,64), vT(B,H,64,NPAD) in bf16.
__global__ __launch_bounds__(256) void k_qkv_gemm(
    const __bf16* __restrict__ xb, const __bf16* __restrict__ wT,
    const float* __restrict__ bias,
    __bf16* __restrict__ qb, __bf16* __restrict__ kb, __bf16* __restrict__ vT) {
  const int lane = threadIdx.x & 31, wave = threadIdx.x >> 5;
  const int lm = lane & 15, lh = lane >> 4;
  const int NT2 = (THEADS * DD) / 32;  // 32
  int job = blockIdx.x * 8 + wave;
  if (job >= (MT788 / 2) * NT2) return;
  int mt2 = job / NT2, nt2 = job - (job / NT2) * NT2;
  const __bf16* arow0 = xb + (size_t)(mt2 * 32 + lm) * CC;
  const __bf16* arow1 = arow0 + (size_t)16 * CC;
  const __bf16* brow0 = wT + (size_t)(nt2 * 32 + lm) * CC;
  const __bf16* brow1 = brow0 + (size_t)16 * CC;
  v8f acc[2][2] = {};
  for (int kc = 0; kc < CC / 32; ++kc) {
    FragA a0, a1, b0, b1;
    load_fragA(a0, arow0 + kc * 32, lh);
    load_fragA(a1, arow1 + kc * 32, lh);
    load_fragB(b0, brow0 + kc * 32, lh);
    load_fragB(b1, brow1 + kc * 32, lh);
    acc[0][0] = wmma_bf16(a0.v, b0.v, acc[0][0]);
    acc[0][1] = wmma_bf16(a0.v, b1.v, acc[0][1]);
    acc[1][0] = wmma_bf16(a1.v, b0.v, acc[1][0]);
    acc[1][1] = wmma_bf16(a1.v, b1.v, acc[1][1]);
  }
#pragma unroll
  for (int tj = 0; tj < 2; ++tj) {
    int col = nt2 * 32 + tj * 16 + lm;  // 0..1023
    int t = col >> 6, d = col & 63;
    float bv = bias[col];
#pragma unroll
    for (int ti = 0; ti < 2; ++ti) {
#pragma unroll
      for (int i = 0; i < 8; ++i) {
        int grow = mt2 * 32 + ti * 16 + i + (lh ? 8 : 0);
        int bb = grow / NN, nn = grow - bb * NN;
        __bf16 val = (__bf16)(acc[ti][tj][i] + bv);
        if (t < 2)
          qb[(((size_t)bb * GHN + t) * NN + nn) * DD + d] = val;
        else if (t < 4)
          kb[(((size_t)bb * GHN + (t - 2)) * NN + nn) * DD + d] = val;
        else
          vT[(((size_t)bb * HH + (t - 4)) * DD + d) * NPAD + nn] = val;
      }
    }
  }
}

// ---------------- fused attention (per batch, per 16-row tile) ---------------
__global__ __launch_bounds__(256) void k_attention(
    const __bf16* __restrict__ qb, const __bf16* __restrict__ kb,
    const __bf16* __restrict__ vT, const float* __restrict__ mw,
    const float* __restrict__ hpw, const float* __restrict__ hpb,
    __bf16* __restrict__ aout) {
  __shared__ float  attn_s[GHN][16][208];   // 26.6 KB
  __shared__ __bf16 p_s[16][HH][NPAD];      // 86   KB
  __shared__ float  inv_s[16][HH];
  __shared__ float  w_s[HH * HH];
  __shared__ float  b_s[HH];

  const int tid = threadIdx.x;
  const int lane = tid & 31, wave = tid >> 5;
  const int lm = lane & 15, lh = lane >> 4;
  const int b  = blockIdx.x / MTILES;
  const int nt = blockIdx.x % MTILES;

  if (tid < HH * HH) w_s[tid] = hpw[tid];
  if (tid < HH) b_s[tid] = hpb[tid];

  // warm L2/WGP caches for data consumed later (global_prefetch_b8)
  {
    const char* vbase = (const char*)(vT + (size_t)b * HH * DD * NPAD);
    const size_t vbytes = (size_t)HH * DD * NPAD * 2;
    for (size_t off = (size_t)tid * 256; off < vbytes; off += 256 * 256)
      __builtin_prefetch(vbase + off, 0, 1);
    int ng0 = nt * 16; if (ng0 > NN - 16) ng0 = NN - 16;
    const char* mbase = (const char*)(mw + (size_t)ng0 * NN * HH);
    const size_t mbytes = (size_t)16 * NN * HH * 4;
    for (size_t off = (size_t)tid * 256; off < mbytes; off += 256 * 256)
      __builtin_prefetch(mbase + off, 0, 1);
  }

  // ---- stage 1: attn = q @ k^T via WMMA ----
  FragA qf[GHN][2];
  {
    int nrow = nt * 16 + lm; if (nrow > NN - 1) nrow = NN - 1;
#pragma unroll
    for (int g = 0; g < GHN; ++g) {
      const __bf16* qrow = qb + (((size_t)b * GHN + g) * NN + nrow) * DD;
#pragma unroll
      for (int kc = 0; kc < 2; ++kc) load_fragA(qf[g][kc], qrow + kc * 32, lh);
    }
  }
  for (int j = wave; j < GHN * MTILES; j += 8) {
    int g = j / MTILES, mt = j % MTILES;
    int mrow = mt * 16 + lm; if (mrow > NN - 1) mrow = NN - 1;
    const __bf16* krow = kb + (((size_t)b * GHN + g) * NN + mrow) * DD;
    v8f acc = {};
#pragma unroll
    for (int kc = 0; kc < 2; ++kc) {
      FragA kf;
      load_fragB(kf, krow + kc * 32, lh);
      acc = wmma_bf16(qf[g][kc].v, kf.v, acc);
    }
#pragma unroll
    for (int i = 0; i < 8; ++i)
      attn_s[g][i + (lh ? 8 : 0)][mt * 16 + lm] = acc[i];
  }
  __syncthreads();

  // ---- stage 2: mask modulate + ReLU + 12x12 head projection ----
  // two adjacent m per thread -> packed b32 LDS stores
  {
    const int PAIRS = (NN + 1) / 2;  // 99
    for (int idx = tid; idx < 16 * PAIRS; idx += 256) {
      int n = idx / PAIRS, mp = idx - (idx / PAIRS) * PAIRS;
      int ng = nt * 16 + n;
      if (ng >= NN) continue;
      int m0 = 2 * mp, m1 = m0 + 1;
      int m1c = (m1 < NN) ? m1 : m0;            // clamp (padding fixed in stage 3)
      float a00 = attn_s[0][n][m0] * SCALE, a10 = attn_s[1][n][m0] * SCALE;
      float a01 = attn_s[0][n][m1c] * SCALE, a11 = attn_s[1][n][m1c] * SCALE;
      const float* mwp0 = mw + ((size_t)ng * NN + m0) * HH;
      const float* mwp1 = mw + ((size_t)ng * NN + m1c) * HH;
      float t0[HH], t1[HH];
#pragma unroll
      for (int h2 = 0; h2 < HH; ++h2) {
        float v0 = (h2 < HRN ? a00 : a10) * mwp0[h2];
        float v1 = (h2 < HRN ? a01 : a11) * mwp1[h2];
        t0[h2] = v0 > 0.f ? v0 : 0.f;
        t1[h2] = v1 > 0.f ? v1 : 0.f;
      }
#pragma unroll
      for (int h = 0; h < HH; ++h) {
        float s0 = b_s[h], s1 = b_s[h];
#pragma unroll
        for (int h2 = 0; h2 < HH; ++h2) {
          float w = w_s[h2 * HH + h];
          s0 += t0[h2] * w;
          s1 += t1[h2] * w;
        }
        union { __bf16 bv[2]; unsigned u; } pk;
        pk.bv[0] = (__bf16)s0;
        pk.bv[1] = (__bf16)s1;
        *(unsigned*)&p_s[n][h][m0] = pk.u;  // m0 even -> 4B aligned
      }
    }
  }
  __syncthreads();

  // ---- stage 3: wave-parallel softmax over m (unnormalized exp, keep 1/sum) -
  for (int pair = wave; pair < 16 * HH; pair += 8) {
    int n = pair / HH, h = pair - (pair / HH) * HH;
    bool valid = (nt * 16 + n) < NN;
    float vals[NPAD / 32];
    float mx = -1e30f;
#pragma unroll
    for (int t = 0; t < NPAD / 32; ++t) {
      int m = lane + 32 * t;
      float v = (valid && m < NN) ? (float)p_s[n][h][m] : -1e30f;
      vals[t] = v;
      mx = v > mx ? v : mx;
    }
#pragma unroll
    for (int off = 16; off; off >>= 1) {
      float o = __shfl_xor(mx, off, 32);
      mx = o > mx ? o : mx;
    }
    float sum = 0.f;
    float ex[NPAD / 32];
#pragma unroll
    for (int t = 0; t < NPAD / 32; ++t) {
      int m = lane + 32 * t;
      ex[t] = (valid && m < NN) ? __expf(vals[t] - mx) : 0.f;
      sum += ex[t];
    }
#pragma unroll
    for (int off = 16; off; off >>= 1) sum += __shfl_xor(sum, off, 32);
#pragma unroll
    for (int t = 0; t < NPAD / 32; ++t)
      p_s[n][h][lane + 32 * t] = (__bf16)ex[t];  // zeros padding too
    if (lane == 0) inv_s[n][h] = valid ? 1.f / sum : 0.f;
  }
  __syncthreads();

  // ---- stage 4: out = softmax(a) @ v via WMMA (A from LDS, B = vT global) ---
  for (int j = wave; j < HH * 4; j += 8) {
    int h = j >> 2, dt = j & 3;
    const __bf16* vrow = vT + (((size_t)b * HH + h) * DD + dt * 16 + lm) * NPAD;
    const __bf16* prow = &p_s[lm][h][0];
    v8f acc = {};
#pragma unroll
    for (int kc = 0; kc < NPAD / 32; ++kc) {
      FragA a, bf;
      load_fragA(a, prow + kc * 32, lh);
      load_fragB(bf, vrow + kc * 32, lh);
      acc = wmma_bf16(a.v, bf.v, acc);
    }
#pragma unroll
    for (int i = 0; i < 8; ++i) {
      int r = i + (lh ? 8 : 0);
      int ng = nt * 16 + r;
      if (ng < NN)
        aout[((size_t)b * NN + ng) * CC + h * DD + dt * 16 + lm] =
            (__bf16)(acc[i] * inv_s[r][h]);
    }
  }
}

// ---------------- output projection: (12608 x 768) @ (768 x 768) + bias -----
// 2x2 register blocking, fp32 output.
__global__ __launch_bounds__(256) void k_proj_gemm(
    const __bf16* __restrict__ ab, const __bf16* __restrict__ wT,
    const float* __restrict__ bias, float* __restrict__ out) {
  const int lane = threadIdx.x & 31, wave = threadIdx.x >> 5;
  const int lm = lane & 15, lh = lane >> 4;
  const int NT2 = CC / 32;  // 24
  int job = blockIdx.x * 8 + wave;
  if (job >= (MT788 / 2) * NT2) return;
  int mt2 = job / NT2, nt2 = job - (job / NT2) * NT2;
  const __bf16* arow0 = ab + (size_t)(mt2 * 32 + lm) * CC;
  const __bf16* arow1 = arow0 + (size_t)16 * CC;
  const __bf16* brow0 = wT + (size_t)(nt2 * 32 + lm) * CC;
  const __bf16* brow1 = brow0 + (size_t)16 * CC;
  v8f acc[2][2] = {};
  for (int kc = 0; kc < CC / 32; ++kc) {
    FragA a0, a1, b0, b1;
    load_fragA(a0, arow0 + kc * 32, lh);
    load_fragA(a1, arow1 + kc * 32, lh);
    load_fragB(b0, brow0 + kc * 32, lh);
    load_fragB(b1, brow1 + kc * 32, lh);
    acc[0][0] = wmma_bf16(a0.v, b0.v, acc[0][0]);
    acc[0][1] = wmma_bf16(a0.v, b1.v, acc[0][1]);
    acc[1][0] = wmma_bf16(a1.v, b0.v, acc[1][0]);
    acc[1][1] = wmma_bf16(a1.v, b1.v, acc[1][1]);
  }
#pragma unroll
  for (int tj = 0; tj < 2; ++tj) {
    int col = nt2 * 32 + tj * 16 + lm;
    float bv = bias[col];
#pragma unroll
    for (int ti = 0; ti < 2; ++ti) {
#pragma unroll
      for (int i = 0; i < 8; ++i) {
        int row = mt2 * 32 + ti * 16 + i + (lh ? 8 : 0);
        out[(size_t)row * CC + col] = acc[ti][tj][i] + bv;
      }
    }
  }
}

// ---------------- host side ----------------
extern "C" void kernel_launch(void* const* d_in, const int* in_sizes, int n_in,
                              void* d_out, int out_size, void* d_ws, size_t ws_size,
                              hipStream_t stream) {
  (void)in_sizes; (void)n_in; (void)out_size; (void)ws_size;
  const float* x       = (const float*)d_in[0];
  const float* qkv_w   = (const float*)d_in[1];
  const float* qkv_b   = (const float*)d_in[2];
  const float* masks   = (const float*)d_in[3];
  const float* mask_pj = (const float*)d_in[4];
  const float* mask_b  = (const float*)d_in[5];
  const float* hp_w    = (const float*)d_in[6];
  const float* hp_b    = (const float*)d_in[7];
  const float* proj_w  = (const float*)d_in[8];
  const float* proj_b  = (const float*)d_in[9];
  float* out = (float*)d_out;

  char* ws = (char*)d_ws;
  size_t o = 0;
  auto alloc = [&](size_t bytes) -> void* {
    o = (o + 255) & ~(size_t)255;
    void* p = ws + o;
    o += bytes;
    return p;
  };

  __bf16* xb     = (__bf16*)alloc((size_t)MROWS * CC * 2);            // 19.4 MB
  __bf16* qkvwT  = (__bf16*)alloc((size_t)(THEADS * DD) * CC * 2);    //  1.6 MB
  __bf16* projwT = (__bf16*)alloc((size_t)CC * CC * 2);               //  1.2 MB
  float*  mwbuf  = (float*) alloc((size_t)NN * NN * HH * 4);          //  1.9 MB
  __bf16* qbuf   = (__bf16*)alloc((size_t)BB * GHN * NN * DD * 2);    //  3.2 MB
  __bf16* kbuf   = (__bf16*)alloc((size_t)BB * GHN * NN * DD * 2);    //  3.2 MB
  __bf16* vTbuf  = (__bf16*)alloc((size_t)BB * HH * DD * NPAD * 2);   // 22.0 MB
  __bf16* aout   = (__bf16*)alloc((size_t)MROWS * CC * 2);            // 19.4 MB

  // prep
  {
    int n = MROWS * CC;
    k_f32_to_bf16<<<(n + 255) / 256, 256, 0, stream>>>(x, xb, n);
  }
  {
    int n = CC * THEADS * DD;
    k_transpose_bf16<<<(n + 255) / 256, 256, 0, stream>>>(qkv_w, qkvwT, CC, THEADS * DD);
  }
  {
    int n = CC * CC;
    k_transpose_bf16<<<(n + 255) / 256, 256, 0, stream>>>(proj_w, projwT, CC, CC);
  }
  {
    int n = NN * NN;
    k_maskw<<<(n + 255) / 256, 256, 0, stream>>>(masks, mask_pj, mask_b, mwbuf);
  }

  // QKV GEMM: 394*32 wave-tiles (32x32 each), 8 waves/block
  k_qkv_gemm<<<((MT788 / 2) * ((THEADS * DD) / 32) + 7) / 8, 256, 0, stream>>>(
      xb, qkvwT, qkv_b, qbuf, kbuf, vTbuf);

  // fused attention: one block per (batch, 16-row tile)
  k_attention<<<BB * MTILES, 256, 0, stream>>>(qbuf, kbuf, vTbuf, mwbuf,
                                               hp_w, hp_b, aout);

  // output projection: 394*24 wave-tiles (32x32 each)
  k_proj_gemm<<<((MT788 / 2) * (CC / 32) + 7) / 8, 256, 0, stream>>>(
      aout, projwT, proj_b, out);
}
